// NeuronMemory_64278480552598
// MI455X (gfx1250) — matrix operand order repaired and checked by reading, not compiled, not run on hardware
//
#include <hip/hip_runtime.h>
#include <hip/hip_bf16.h>
#include <string.h>

// ---------------------------------------------------------------------------
// NeuronMemory fused retrieval for MI455X (gfx1250, wave32, WMMA)
//
// Router GEMM x[4096,1024] @ W[1024,32768] = 274.9 GFLOP dominates (>99.8%).
// W_router is converted once to bf16 (64MB -> resident in the 192MB L2);
// 128 persistent workgroups (32 tokens each) sweep all of N with
// v_wmma_f32_16x16x32_bf16, sharing each B fragment across TWO WMMAs
// (two 16-token A tiles) to halve L2 operand bandwidth, and maintain an
// exact streaming top-64 in LDS so logits never touch global memory.
// Stage 2 (query/fine-score/top-16/softmax/V-combine, <0.5 GFLOP) is fused.
// ---------------------------------------------------------------------------

#define D_MODEL    1024
#define N_KNOW     32768
#define RANK       128
#define K_RANK     128
#define COARSE_K   64
#define FINE_K     16
#define N_TOKENS   4096          // B*S
#define TOK_PER_WG 32
#define NEG_INF    (-3.402823466e38f)

typedef __attribute__((ext_vector_type(16))) __bf16 v16bf;
typedef __attribute__((ext_vector_type(8)))  float  v8f;

union AFrag { uint4 q[2]; v16bf v; };   // 32B: 16 bf16 WMMA operand

// ---- dynamic LDS layout (bytes) -------------------------------------------
// stage 1:
//   [0      , 65536) bf16 A tile  xa[32][1024]
//   [65536  , 81920) float cval[32][128]
//   [81920  , 98304) int   cidx[32][128]
//   [98304  ,106496) float tval[32][64]
//   [106496 ,114688) int   tidxl[32][64]
//   [114688 ,114816) float thr[32]
//   [114816 ,114944) int   fillc[32]
//   [114944 ,115072) int   ccnt[32]
// stage 2 overlays the A-tile region:
//   [0,16384) hld[32][128]  [16384,32768) qry[32][128]
//   [32768,40960) fsv[32][64]  [40960,43008) fw[32][16]  [43008,45056) fgi
#define SMEM_BYTES 115072

static __device__ __forceinline__ unsigned short f2bf(float f) {
    __bf16 h = (__bf16)f;               // RNE convert
    return __builtin_bit_cast(unsigned short, h);
}
static __device__ __forceinline__ unsigned pack2(float lo, float hi) {
    return (unsigned)f2bf(lo) | ((unsigned)f2bf(hi) << 16);
}

// ---------------------------------------------------------------------------
// Kernel 1: W_router f32 -> bf16 (row-major [K=1024][N=32768]), 8 elems/thread
// ---------------------------------------------------------------------------
__global__ __launch_bounds__(256) void cvt_w_bf16(const float* __restrict__ W,
                                                  unsigned short* __restrict__ Wb) {
    size_t i = ((size_t)blockIdx.x * 256 + threadIdx.x) * 8;
    float4 a = *(const float4*)(W + i);
    float4 b = *(const float4*)(W + i + 4);
    uint4 o;
    o.x = pack2(a.x, a.y);
    o.y = pack2(a.z, a.w);
    o.z = pack2(b.x, b.y);
    o.w = pack2(b.z, b.w);
    *(uint4*)(Wb + i) = o;              // 16B coalesced store
}

// ---------------------------------------------------------------------------
// Kernel 2: fused router GEMM + top-64 + fine match + combine.
// 256 threads = 8 wave32 waves; WG owns 32 tokens; each wave owns 16 N-columns
// per tile and issues 2 WMMAs per B fragment (tokens 0-15 and 16-31).
// ---------------------------------------------------------------------------
__global__ __launch_bounds__(256) void neuron_memory_fused(
    const float* __restrict__ x,             // [4096,1024]
    const float* __restrict__ h,             // [4096,128]
    const unsigned short* __restrict__ Wb,   // bf16 [1024,32768]
    const float* __restrict__ Wq,            // [128,128]
    const float* __restrict__ Kall,          // [32768,128]
    const float* __restrict__ Vall,          // [32768,1024]
    float* __restrict__ out)                 // [4096,1024]
{
    extern __shared__ char smem[];
    __bf16* xa   = (__bf16*)smem;
    float*  cval = (float*)(smem + 65536);
    int*    cidx = (int*)  (smem + 81920);
    float*  tval = (float*)(smem + 98304);
    int*    tidxl= (int*)  (smem + 106496);
    float*  thr  = (float*)(smem + 114688);
    int*    fillc= (int*)  (smem + 114816);
    int*    ccnt = (int*)  (smem + 114944);
    // stage-2 overlay
    float*  hld  = (float*)smem;             // [32][128]
    float*  qry  = (float*)(smem + 16384);   // [32][128]
    float*  fsv  = (float*)(smem + 32768);   // [32][64]
    float*  fw   = (float*)(smem + 40960);   // [32][16]
    int*    fgi  = (int*)  (smem + 43008);   // [32][16]

    const int tid  = threadIdx.x;
    const int wave = tid >> 5;               // wave32: 8 waves
    const int lane = tid & 31;
    const int tok0 = blockIdx.x * TOK_PER_WG;

    // ---- stage A operand: 32 x-rows -> bf16 LDS ----
    for (int i = tid; i < TOK_PER_WG * D_MODEL; i += 256) {
        int t = i >> 10, d = i & (D_MODEL - 1);
        xa[i] = (__bf16)x[(size_t)(tok0 + t) * D_MODEL + d];
    }
    if (tid < TOK_PER_WG) { thr[tid] = NEG_INF; fillc[tid] = 0; ccnt[tid] = 0; }
    __syncthreads();

    // ISA 16-bit A layout: lane%16 = M row; lanes 0-15 carry K {0..7,16..23},
    // lanes 16-31 carry K {8..15,24..31}.  abase1 = second 16-token tile.
    const int abase0 = ((lane & 15) << 10) + ((lane >> 4) << 3);
    const int abase1 = abase0 + (16 << 10);
    // B layout: lane%16 = N column; lanes 0-15 carry K {0..7,16..23}, lanes 16-31 K {8..15,24..31}.
    const unsigned short* bp0 = Wb + (size_t)((lane >> 4) << 3) * N_KNOW + (lane & 15);

    // ================= stage 1: GEMM + streaming top-64 =================
    for (int n0 = 0; n0 < N_KNOW; n0 += 128) {
        const int ncol = n0 + wave * 16;
        const unsigned short* bp = bp0 + ncol;

        v8f c0 = {}, c1 = {};
#pragma unroll 4
        for (int k0 = 0; k0 < D_MODEL; k0 += 32) {
            AFrag a0, a1, b;
            const unsigned short* bk = bp + (size_t)k0 * N_KNOW;
            for (int s = 0; s < 16; ++s)
                b.v[s] = __builtin_bit_cast(__bf16, bk[(size_t)((s & 7) + ((s >> 3) << 4)) * N_KNOW]);
            a0.q[0] = *(const uint4*)&xa[abase0 + k0];            // ds loads
            a0.q[1] = *(const uint4*)&xa[abase0 + k0 + 16];
            a1.q[0] = *(const uint4*)&xa[abase1 + k0];
            a1.q[1] = *(const uint4*)&xa[abase1 + k0 + 16];
            c0 = __builtin_amdgcn_wmma_f32_16x16x32_bf16(
                     false, a0.v, false, b.v, (short)0, c0, false, false);
            c1 = __builtin_amdgcn_wmma_f32_16x16x32_bf16(
                     false, a1.v, false, b.v, (short)0, c1, false, false);
        }

        // pull this lane's 16 thresholds once (4x ds_load_b128)
        const int tb = (lane >> 4) << 3;          // token base for c0 rows
        float4 ta0 = *(const float4*)&thr[tb];
        float4 ta1 = *(const float4*)&thr[tb + 4];
        float4 tb0 = *(const float4*)&thr[tb + 16];
        float4 tb1 = *(const float4*)&thr[tb + 20];
        float thr0[8] = { ta0.x, ta0.y, ta0.z, ta0.w, ta1.x, ta1.y, ta1.z, ta1.w };
        float thr1[8] = { tb0.x, tb0.y, tb0.z, tb0.w, tb1.x, tb1.y, tb1.z, tb1.w };

        // C layout: VGPR r -> (M=r, N=lane) lanes 0-15 ; (M=8+r, N=lane-16)
        const int nglob = ncol + (lane & 15);
#pragma unroll
        for (int r = 0; r < 8; ++r) {
            int t0 = tb + r;
            float v0 = c0[r];
            if (v0 > thr0[r]) {
                int p = atomicAdd(&ccnt[t0], 1);
                if (p < 128) { cval[t0 * 128 + p] = v0; cidx[t0 * 128 + p] = nglob; }
            }
            int t1 = t0 + 16;
            float v1 = c1[r];
            if (v1 > thr1[r]) {
                int p = atomicAdd(&ccnt[t1], 1);
                if (p < 128) { cval[t1 * 128 + p] = v1; cidx[t1 * 128 + p] = nglob; }
            }
        }
        __syncthreads();

        // token-owner threads merge staged candidates into running top-64
        if (tid < TOK_PER_WG) {
            int t = tid;
            int nc = ccnt[t]; if (nc > 128) nc = 128;
            int f = fillc[t]; float th = thr[t];
            float* tv = tval + t * COARSE_K;
            int*   ti = tidxl + t * COARSE_K;
            for (int i = 0; i < nc; ++i) {
                float v = cval[t * 128 + i]; int ix = cidx[t * 128 + i];
                if (f < COARSE_K) {
                    tv[f] = v; ti[f] = ix; ++f;
                    if (f == COARSE_K) {
                        th = tv[0];
                        for (int j = 1; j < COARSE_K; ++j) th = fminf(th, tv[j]);
                    }
                } else if (v > th) {
                    int mp = 0; float mv = tv[0];
                    for (int j = 1; j < COARSE_K; ++j)
                        if (tv[j] < mv) { mv = tv[j]; mp = j; }
                    tv[mp] = v; ti[mp] = ix;
                    th = tv[0];
                    for (int j = 1; j < COARSE_K; ++j) th = fminf(th, tv[j]);
                }
            }
            fillc[t] = f; thr[t] = (f == COARSE_K) ? th : NEG_INF; ccnt[t] = 0;
        }
        __syncthreads();
    }

    // ================= stage 2: fine matching (tiny) =================
    for (int i = tid; i < TOK_PER_WG * RANK; i += 256)
        hld[i] = h[(size_t)(tok0 + (i >> 7)) * RANK + (i & (RANK - 1))];
    __syncthreads();

    // query[t][k] = sum_r h[t][r] * Wq[k][r]   (8 threads per token, 16 k each)
    {
        int t = tid >> 3, kb = (tid & 7) * 16;
        const float* hr = hld + t * RANK;
        for (int k = kb; k < kb + 16; ++k) {
            const float* wq = Wq + (size_t)k * RANK;
            float s = 0.f;
#pragma unroll 8
            for (int r = 0; r < RANK; ++r) s += hr[r] * wq[r];
            qry[t * K_RANK + k] = s;
        }
    }
    __syncthreads();

    // fine scores over the 64 coarse candidates (8 threads/token, 8 cands each)
    {
        int t = tid >> 3, cb = (tid & 7) * 8;
        const float* q = qry + t * K_RANK;
        for (int cI = cb; cI < cb + 8; ++cI) {
            const float* kr = Kall + (size_t)tidxl[t * COARSE_K + cI] * K_RANK;
            float s = 0.f;
#pragma unroll 8
            for (int r = 0; r < K_RANK; ++r) s += q[r] * kr[r];
            fsv[t * COARSE_K + cI] = s * 0.08838834764831845f;   // 1/sqrt(128)
        }
    }
    __syncthreads();

    // top-16 + softmax per token (owner thread)
    if (tid < TOK_PER_WG) {
        int t = tid;
        float vals[FINE_K]; int gix[FINE_K];
        for (int f = 0; f < FINE_K; ++f) {
            int mp = 0; float mv = NEG_INF;
            for (int j = 0; j < COARSE_K; ++j) {
                float v = fsv[t * COARSE_K + j];
                if (v > mv) { mv = v; mp = j; }
            }
            vals[f] = mv; gix[f] = tidxl[t * COARSE_K + mp];
            fsv[t * COARSE_K + mp] = NEG_INF;
        }
        float mx = vals[0], se = 0.f;
        for (int f = 0; f < FINE_K; ++f) { vals[f] = __expf(vals[f] - mx); se += vals[f]; }
        float inv = 1.f / se;
        for (int f = 0; f < FINE_K; ++f) { fw[t * FINE_K + f] = vals[f] * inv;
                                           fgi[t * FINE_K + f] = gix[f]; }
    }
    __syncthreads();

    // weighted V combine: out[t][d] = sum_f w*V[gidx][d]; coalesced over d
    for (int t = 0; t < TOK_PER_WG; ++t) {
        float acc0 = 0.f, acc1 = 0.f, acc2 = 0.f, acc3 = 0.f;
#pragma unroll
        for (int f = 0; f < FINE_K; ++f) {
            float w = fw[t * FINE_K + f];
            const float* vr = Vall + (size_t)fgi[t * FINE_K + f] * D_MODEL;
            acc0 += w * vr[tid];
            acc1 += w * vr[tid + 256];
            acc2 += w * vr[tid + 512];
            acc3 += w * vr[tid + 768];
        }
        float* op = out + (size_t)(tok0 + t) * D_MODEL;
        op[tid]       = acc0;
        op[tid + 256] = acc1;
        op[tid + 512] = acc2;
        op[tid + 768] = acc3;
    }
}

// ---------------------------------------------------------------------------
extern "C" void kernel_launch(void* const* d_in, const int* in_sizes, int n_in,
                              void* d_out, int out_size, void* d_ws, size_t ws_size,
                              hipStream_t stream) {
    const float* x    = (const float*)d_in[0];
    const float* h    = (const float*)d_in[1];
    const float* Wr   = (const float*)d_in[2];
    const float* Wq   = (const float*)d_in[3];
    const float* Kall = (const float*)d_in[4];
    const float* Vall = (const float*)d_in[5];
    float* out = (float*)d_out;

    // ws: bf16 copy of W_router, 1024*32768*2 = 64 MB (L2-resident on MI455X)
    unsigned short* Wb = (unsigned short*)d_ws;

    // allow >64KB dynamic LDS (gfx1250 WGP has 320KB)
    (void)hipFuncSetAttribute(reinterpret_cast<const void*>(neuron_memory_fused),
                              hipFuncAttributeMaxDynamicSharedMemorySize, SMEM_BYTES);

    const int nW = D_MODEL * N_KNOW;                    // 33,554,432 elems
    cvt_w_bf16<<<nW / (256 * 8), 256, 0, stream>>>(Wr, Wb);

    neuron_memory_fused<<<N_TOKENS / TOK_PER_WG, 256, SMEM_BYTES, stream>>>(
        x, h, Wb, Wq, Kall, Vall, out);
}
